// LSTMBlock_41480794145354
// MI455X (gfx1250) — compile-verified, matrix operand order checked by reading, not run on hardware
//
#include <hip/hip_runtime.h>
#include <math.h>

#define BB 256
#define II 256
#define HH 512
#define TT 400
#define KT_X 8     // 256/32 k-tiles from x
#define KT_H 16    // 512/32 k-tiles from h
#define KT_ALL 24
#define KTILE_Q 64 // one k-tile (32 lanes x 16 bf16) = 1024 B = 64 uint4

typedef __attribute__((ext_vector_type(16))) __bf16 v16bf;
typedef __attribute__((ext_vector_type(8)))  float  v8f;

union Frag {
    v16bf v;
    uint4 q[2];
};

__device__ __forceinline__ unsigned short f2bf(float f) {
    unsigned u = __float_as_uint(f);
    u += 0x7FFFu + ((u >> 16) & 1u);   // round-to-nearest-even
    return (unsigned short)(u >> 16);
}

// K offset inside a 16x32 A/B fragment for (pair j, lane-half): §7.12.2 layout
__device__ __forceinline__ unsigned koff(unsigned j, unsigned half) {
    return ((j < 4u) ? 2u * j : 16u + 2u * (j - 4u)) + 8u * half;
}

// x packed directly into WMMA A-fragment layout:
// xpk[t(400)][mt(16)][kt(8)][lane(32)][e(16)]
__global__ void lstm_pack_x(const float* __restrict__ x,
                            unsigned short* __restrict__ xpk) {
    unsigned o = blockIdx.x * 256u + threadIdx.x;  // 400*16*8*32*16 = 26,214,400
    unsigned e    = o & 15u;
    unsigned lane = (o >> 4) & 31u;
    unsigned kt   = (o >> 9) & 7u;
    unsigned mt   = (o >> 12) & 15u;
    unsigned t    = o >> 16;
    unsigned j = e >> 1, p = e & 1u, half = lane >> 4;
    unsigned b = mt * 16u + (lane & 15u);
    unsigned k = kt * 32u + koff(j, half) + p;
    xpk[o] = f2bf(x[(b * II + k) * TT + t]);
}

// Combined [w_ih | w_hh] packed into WMMA B-operand tile layout:
// wpk[nt(128)][kt(24)][lane(32)][e(16)], B[k][n] = w[n][k]
__global__ void lstm_pack_w(const float* __restrict__ w_ih,
                            const float* __restrict__ w_hh,
                            unsigned short* __restrict__ wpk) {
    unsigned o = blockIdx.x * 256u + threadIdx.x;  // 128*24*32*16 = 1,572,864
    unsigned e    = o & 15u;
    unsigned lane = (o >> 4) & 31u;
    unsigned kt   = (o >> 9) % KT_ALL;
    unsigned nt   = o / (KT_ALL * 512u);
    unsigned j = e >> 1, p = e & 1u, half = lane >> 4;
    unsigned n  = nt * 16u + (lane & 15u);
    unsigned kk = koff(j, half) + p;
    float v;
    if (kt < KT_X) v = w_ih[n * II + kt * 32u + kk];
    else           v = w_hh[n * HH + (kt - KT_X) * 32u + kk];
    wpk[o] = f2bf(v);
}

#define MMA_BODY()                                                         \
    do {                                                                   \
        Frag aF0, aF1, bF0, bF1;                                           \
        aF0.q[0] = ap0[0]; aF0.q[1] = ap0[1]; ap0 += KTILE_Q;              \
        aF1.q[0] = ap1[0]; aF1.q[1] = ap1[1]; ap1 += KTILE_Q;              \
        bF0.q[0] = bp0[0]; bF0.q[1] = bp0[1]; bp0 += KTILE_Q;              \
        bF1.q[0] = bp1[0]; bF1.q[1] = bp1[1]; bp1 += KTILE_Q;              \
        acc00 = __builtin_amdgcn_wmma_f32_16x16x32_bf16(                   \
            false, aF0.v, false, bF0.v, (short)0, acc00, false, false);    \
        acc01 = __builtin_amdgcn_wmma_f32_16x16x32_bf16(                   \
            false, aF0.v, false, bF1.v, (short)0, acc01, false, false);    \
        acc10 = __builtin_amdgcn_wmma_f32_16x16x32_bf16(                   \
            false, aF1.v, false, bF0.v, (short)0, acc10, false, false);    \
        acc11 = __builtin_amdgcn_wmma_f32_16x16x32_bf16(                   \
            false, aF1.v, false, bF1.v, (short)0, acc11, false, false);    \
    } while (0)

// Fused LSTM step: two straight-line WMMA K-loops (x part, then h part,
// constant-stride pointer walks, no per-iteration branching), one LDS
// exchange of the 4 gate tiles, then activations + cell/hidden update.
// Block = 32 batch rows x 32 h-cols; wave g owns gate g. h is written back
// already in packed A-fragment layout for the next step.
__launch_bounds__(128)
__global__ void lstm_step(const unsigned short* __restrict__ xpk,
                          const unsigned short* __restrict__ wpk,
                          const unsigned short* __restrict__ hpk_in,
                          unsigned short* __restrict__ hpk_out,
                          const float* __restrict__ b_ih,
                          const float* __restrict__ b_hh,
                          float* __restrict__ out,
                          int t) {
    __shared__ float sG[4][32][32];      // i,f,g,o tiles

    const int tid  = threadIdx.x;
    const int lane = tid & 31;
    const int wg   = tid >> 5;           // gate index 0..3
    const int half = lane >> 4;
    const int ln   = lane & 15;
    const int m0 = blockIdx.x * 32;      // batch rows
    const int n0 = blockIdx.y * 32;      // h columns

    v8f zero = {};
    v8f acc00 = zero, acc01 = zero, acc10 = zero, acc11 = zero;

    const int mt0 = blockIdx.x * 2;
    const int ntb = wg * 32 + blockIdx.y * 2;   // col = wg*512 + n0

    // B pointers walk contiguously through all 24 k-tiles (x then h region)
    const uint4* bp0 = reinterpret_cast<const uint4*>(wpk) +
                       (size_t)(ntb + 0) * KT_ALL * KTILE_Q + lane * 2;
    const uint4* bp1 = reinterpret_cast<const uint4*>(wpk) +
                       (size_t)(ntb + 1) * KT_ALL * KTILE_Q + lane * 2;
    // A pointers: x-region first
    const uint4* ap0 = reinterpret_cast<const uint4*>(xpk) +
                       ((size_t)t * 16 + mt0 + 0) * KT_X * KTILE_Q + lane * 2;
    const uint4* ap1 = reinterpret_cast<const uint4*>(xpk) +
                       ((size_t)t * 16 + mt0 + 1) * KT_X * KTILE_Q + lane * 2;

    for (int kt = 0; kt < KT_X; ++kt) {
        MMA_BODY();
    }
    if (t != 0) {                                 // h==0 at t=0: skip h part
        ap0 = reinterpret_cast<const uint4*>(hpk_in) +
              (size_t)(mt0 + 0) * KT_H * KTILE_Q + lane * 2;
        ap1 = reinterpret_cast<const uint4*>(hpk_in) +
              (size_t)(mt0 + 1) * KT_H * KTILE_Q + lane * 2;
        for (int kt = 0; kt < KT_H; ++kt) {
            MMA_BODY();
        }
    }

    // exchange gate tiles through LDS (C/D layout: VGPR e -> M = e + 8*half)
    #pragma unroll
    for (int e = 0; e < 8; ++e) {
        int r0 = e + 8 * half;
        sG[wg][r0     ][ln     ] = acc00[e];
        sG[wg][r0     ][16 + ln] = acc01[e];
        sG[wg][16 + r0][ln     ] = acc10[e];
        sG[wg][16 + r0][16 + ln] = acc11[e];
    }
    __syncthreads();

    // pointwise LSTM update: 1024 elements / 128 threads = 8 each
    #pragma unroll
    for (int e = 0; e < 8; ++e) {
        int idx = tid + e * 128;
        int row = idx >> 5, col = idx & 31;
        int bb = m0 + row;
        int hc = n0 + col;
        float iv = sG[0][row][col] + b_ih[hc]          + b_hh[hc];
        float fv = sG[1][row][col] + b_ih[HH + hc]     + b_hh[HH + hc];
        float gv = sG[2][row][col] + b_ih[2 * HH + hc] + b_hh[2 * HH + hc];
        float ov = sG[3][row][col] + b_ih[3 * HH + hc] + b_hh[3 * HH + hc];
        iv = 1.f / (1.f + __expf(-iv));
        fv = 1.f / (1.f + __expf(-fv));
        gv = tanhf(gv);
        ov = 1.f / (1.f + __expf(-ov));
        float cp = (t == 0) ? 0.f : out[((size_t)(t - 1) * BB + bb) * HH + hc];
        float cn = fv * cp + iv * gv;
        out[((size_t)t * BB + bb) * HH + hc] = cn;
        float hn = ov * tanhf(cn);
        // scatter h into packed A-fragment layout for next step
        int mt  = bb >> 4, r = bb & 15;
        int kt2 = hc >> 5, kk = hc & 31;
        int hf  = (kk >> 3) & 1;
        int jj  = ((kk >> 4) & 1) * 4 + ((kk & 7) >> 1);
        int pp  = kk & 1;
        hpk_out[(((size_t)(mt * KT_H + kt2)) * 32 + (r + 16 * hf)) * 16 + 2 * jj + pp]
            = f2bf(hn);
    }
}

extern "C" void kernel_launch(void* const* d_in, const int* in_sizes, int n_in,
                              void* d_out, int out_size, void* d_ws, size_t ws_size,
                              hipStream_t stream) {
    const float* x    = (const float*)d_in[0];
    const float* w_ih = (const float*)d_in[1];
    const float* w_hh = (const float*)d_in[2];
    const float* b_ih = (const float*)d_in[3];
    const float* b_hh = (const float*)d_in[4];
    float* out = (float*)d_out;

    // workspace layout (~56 MB)
    char* ws = (char*)d_ws;
    unsigned short* xpk = (unsigned short*)ws;                        // 52,428,800 B
    unsigned short* wpk = (unsigned short*)(ws + 52428800);           //  3,145,728 B
    unsigned short* h0  = (unsigned short*)(ws + 52428800 + 3145728); //    262,144 B
    unsigned short* h1  = h0 + (size_t)BB * HH;                       //    262,144 B

    lstm_pack_x<<<(TT * BB * II) / 256, 256, 0, stream>>>(x, xpk);
    lstm_pack_w<<<(128 * KT_ALL * 32 * 16) / 256, 256, 0, stream>>>(w_ih, w_hh, wpk);

    unsigned short* hb[2] = {h0, h1};
    for (int t = 0; t < TT; ++t) {
        lstm_step<<<dim3(BB / 32, HH / 32), 128, 0, stream>>>(
            xpk, wpk, hb[t & 1], hb[(t + 1) & 1], b_ih, b_hh, out, t);
    }
}
